// PerturbedTopK_14577119003149
// MI455X (gfx1250) — 8-bit matrix operand order repaired and checked by reading, compiled, not run on hardware
//
#include <hip/hip_runtime.h>

typedef __attribute__((ext_vector_type(8))) int   v8i;
typedef __attribute__((ext_vector_type(4))) float v4f;

#define NROWS   392
#define NSAMP   500
#define DDIM    256
#define KSEL    32
#define SIGMA_F 0.05f

__device__ __forceinline__ unsigned int ordkey(float f) {
  unsigned int u = __float_as_uint(f);
  return u ^ ((unsigned int)((int)u >> 31) | 0x80000000u);
}

extern "C" __global__ void __launch_bounds__(256)
ptopk_kernel(const float* __restrict__ x, const float* __restrict__ noise,
             float* __restrict__ out) {
  __shared__ unsigned int counts[KSEL * DDIM];
  __shared__ unsigned int maskBufW[(8 * 16 * DDIM) / 4];

  const int row  = blockIdx.x;
  const int tid  = threadIdx.x;
  const int lane = tid & 31;
  const int wave = tid >> 5;
  const int hi   = lane >> 4;
  const int nlo  = lane & 15;

  for (int i = tid; i < KSEL * DDIM; i += 256) counts[i] = 0u;
  __syncthreads();

  const v4f* x4 = (const v4f*)(x + row * DDIM);
  const v4f xa = x4[lane];
  const v4f xb = x4[32 + lane];

  unsigned char* mask = (unsigned char*)maskBufW + wave * (16 * DDIM);
  const float* nrow = noise + (size_t)row * (NSAMP * DDIM);

  for (int batch = wave; batch < 32; batch += 8) {
    for (int s = 0; s < 16; ++s) {
      const int sg = batch * 16 + s;
      unsigned int w0 = 0u, w1 = 0u;
      if (sg < NSAMP) {
        const v4f* n4 = (const v4f*)(nrow + sg * DDIM);
        if (sg + 1 < NSAMP) {
          __builtin_prefetch(nrow + (sg + 1) * DDIM + 4 * lane, 0, 0);
          __builtin_prefetch(nrow + (sg + 1) * DDIM + 128 + 4 * lane, 0, 0);
        }
        const v4f na = __builtin_nontemporal_load(n4 + lane);
        const v4f nb = __builtin_nontemporal_load(n4 + 32 + lane);
        unsigned int u[8];
        u[0] = ordkey(fmaf(na[0], SIGMA_F, xa[0]));
        u[1] = ordkey(fmaf(na[1], SIGMA_F, xa[1]));
        u[2] = ordkey(fmaf(na[2], SIGMA_F, xa[2]));
        u[3] = ordkey(fmaf(na[3], SIGMA_F, xa[3]));
        u[4] = ordkey(fmaf(nb[0], SIGMA_F, xb[0]));
        u[5] = ordkey(fmaf(nb[1], SIGMA_F, xb[1]));
        u[6] = ordkey(fmaf(nb[2], SIGMA_F, xb[2]));
        u[7] = ordkey(fmaf(nb[3], SIGMA_F, xb[3]));

        unsigned int P = 0u, rem = KSEL;
        for (int b = 31; b >= 0; --b) {
          const unsigned int pat = (P >> b) | 1u;
          unsigned int cnt = 0;
#pragma unroll
          for (int e = 0; e < 8; ++e)
            cnt += __builtin_popcount(
                __builtin_amdgcn_ballot_w32((u[e] >> b) == pat));
          if (cnt >= rem) P |= (1u << b); else rem -= cnt;
        }
#pragma unroll
        for (int e = 0; e < 4; ++e) w0 |= (u[e]     >= P ? 1u : 0u) << (8 * e);
#pragma unroll
        for (int e = 0; e < 4; ++e) w1 |= (u[e + 4] >= P ? 1u : 0u) << (8 * e);
      }
      *(unsigned int*)(mask + s * DDIM +       4 * lane) = w0;
      *(unsigned int*)(mask + s * DDIM + 128 + 4 * lane) = w1;
    }

    const unsigned char* mrow = mask + nlo * DDIM;
    v8i A0, A1, A2, A3;
#pragma unroll
    for (int v = 0; v < 8; ++v) {
      const int koff = ((v >> 1) << 4) + ((v & 1) << 2) + (hi << 3);
      A0[v] = *(const int*)(mrow +   0 + koff);
      A1[v] = *(const int*)(mrow +  64 + koff);
      A2[v] = *(const int*)(mrow + 128 + koff);
      A3[v] = *(const int*)(mrow + 192 + koff);
    }
    v8i ones;
#pragma unroll
    for (int v = 0; v < 8; ++v) ones[v] = 0x01010101;

#pragma unroll
    for (int t = 0; t < 16; ++t) {
      v8i acc = {0, 0, 0, 0, 0, 0, 0, 0};
      const int c0 = t >> 2;
      if (c0 > 0) acc = __builtin_amdgcn_wmma_i32_16x16x64_iu8(false, A0, false, ones, acc, false, false);
      if (c0 > 1) acc = __builtin_amdgcn_wmma_i32_16x16x64_iu8(false, A1, false, ones, acc, false, false);
      if (c0 > 2) acc = __builtin_amdgcn_wmma_i32_16x16x64_iu8(false, A2, false, ones, acc, false, false);

      const int d = t * 16 + nlo;
      v8i Bb;
#pragma unroll
      for (int v = 0; v < 8; ++v) {
        const int kbase = ((v >> 1) << 4) + ((v & 1) << 2) + (hi << 3);
        const int diff  = d - (c0 * 64 + kbase);
        const unsigned int val =
            diff >= 4 ? 0x01010101u :
            diff == 3 ? 0x40010101u :
            diff == 2 ? 0x00400101u :
            diff == 1 ? 0x00004001u :
            diff == 0 ? 0x00000040u : 0u;
        Bb[v] = (int)val;
      }
      const v8i Ab = (c0 == 0) ? A0 : (c0 == 1) ? A1 : (c0 == 2) ? A2 : A3;
      acc = __builtin_amdgcn_wmma_i32_16x16x64_iu8(false, Ab, false, Bb, acc, false, false);

#pragma unroll
      for (int r = 0; r < 8; ++r) {
        const int R2 = acc[r];
        if (R2 >= 64) {
          const int j = R2 - 64;
          if (j < KSEL) atomicAdd(&counts[j * DDIM + d], 1u);
        }
      }
    }
  }

  __syncthreads();
  const float inv = 1.0f / (float)NSAMP;
  float* orow = out + (size_t)row * (KSEL * DDIM);
  for (int i = tid; i < KSEL * DDIM; i += 256) orow[i] = (float)counts[i] * inv;
}

extern "C" void kernel_launch(void* const* d_in, const int* in_sizes, int n_in,
                              void* d_out, int out_size, void* d_ws, size_t ws_size,
                              hipStream_t stream) {
  (void)in_sizes; (void)n_in; (void)d_ws; (void)ws_size; (void)out_size;
  const float* x     = (const float*)d_in[0];
  const float* noise = (const float*)d_in[1];
  float* out = (float*)d_out;
  hipLaunchKernelGGL(ptopk_kernel, dim3(NROWS), dim3(256), 0, stream,
                     x, noise, out);
}
